// SpatialMixPeak_91250875171040
// MI455X (gfx1250) — matrix operand order repaired and checked by reading, not compiled, not run on hardware
//
#include <hip/hip_runtime.h>
#include <stdint.h>

// ---------------- problem constants (from setup_inputs) ----------------
constexpr int B_ = 2;
constexpr int H_ = 64;
constexpr int W_ = 64;
constexpr int T_ = H_ * W_;     // 4096
constexpr int C_ = 192;
constexpr int NC_ = 64;         // scan chunks
constexpr int LC_ = 64;         // chunk length  (NC_*LC_ == T_)

typedef __attribute__((ext_vector_type(16))) __bf16 v16bf;
typedef __attribute__((ext_vector_type(8)))  float  v8f;

union FragBF { v16bf bf; uint4 u[2]; };

__device__ __forceinline__ unsigned short f2bf(float f) {
  unsigned u = __float_as_uint(f);
  unsigned r = u + 0x7FFFu + ((u >> 16) & 1u);   // round-to-nearest-even
  return (unsigned short)(r >> 16);
}

// ---------------- kernel 1: q_shift + token mixes -> bf16 ----------------
__global__ void __launch_bounds__(256) k_prep(
    const float* __restrict__ x,
    const float* __restrict__ mk, const float* __restrict__ mv, const float* __restrict__ mr,
    unsigned short* __restrict__ xk, unsigned short* __restrict__ xv, unsigned short* __restrict__ xr) {
  int idx = blockIdx.x * blockDim.x + threadIdx.x;   // B*T*C
  if (idx >= B_ * T_ * C_) return;
  int c = idx % C_;
  int t = (idx / C_) % T_;
  int b = idx / (C_ * T_);
  int h = t >> 6, w = t & 63;
  const float* xb = x + (size_t)b * T_ * C_;
  float xc = xb[(size_t)t * C_ + c];
  float xx = 0.f;
  int g = c / 48;                                     // gamma = 0.25 -> 4 groups of 48
  if (g == 0)      { if (w >= 1)      xx = xb[(size_t)(t - 1)  * C_ + c]; }
  else if (g == 1) { if (w <= W_ - 2) xx = xb[(size_t)(t + 1)  * C_ + c]; }
  else if (g == 2) { if (h >= 1)      xx = xb[(size_t)(t - W_) * C_ + c]; }
  else             { if (h <= H_ - 2) xx = xb[(size_t)(t + W_) * C_ + c]; }
  float a;
  a = mk[c]; xk[idx] = f2bf(xc * a + xx * (1.f - a));
  a = mv[c]; xv[idx] = f2bf(xc * a + xx * (1.f - a));
  a = mr[c]; xr[idx] = f2bf(xc * a + xx * (1.f - a));
}

// -------- kernel 2: weight transpose f32 -> bf16 (N-major for WMMA B) --------
__global__ void __launch_bounds__(256) k_wtrans(
    const float* __restrict__ Wk, const float* __restrict__ Wv,
    const float* __restrict__ Wr, const float* __restrict__ Wo,
    unsigned short* __restrict__ WT) {
  int idx = blockIdx.x * blockDim.x + threadIdx.x;   // 4*C*C
  if (idx >= 4 * C_ * C_) return;
  int mat = idx / (C_ * C_);
  int r   = idx % (C_ * C_);
  int n = r / C_;
  int k = r % C_;
  const float* W = (mat == 0) ? Wk : (mat == 1) ? Wv : (mat == 2) ? Wr : Wo;
  WT[idx] = f2bf(W[(size_t)k * C_ + n]);             // WT[mat][n*C + k]
}

// ---------------- kernel 3: WMMA bf16 GEMM, C(8192x192) = A(8192x192) @ B(192x192) ----------------
// one wave per 16x16 output tile; K = 192 = 6 x 32
__global__ void __launch_bounds__(32) k_gemm(
    const unsigned short* __restrict__ A,   // M x 192 bf16, row-major
    const unsigned short* __restrict__ BT,  // 192 x 192 bf16, N-major (row n holds K contiguous)
    float* __restrict__ Cout, int act) {
  int lane = threadIdx.x;
  int half = lane >> 4;
  int l16  = lane & 15;
  int row  = blockIdx.x * 16 + l16;
  int ncol = blockIdx.y * 16 + l16;
  const unsigned short* arow = A  + (size_t)row  * C_;
  const unsigned short* brow = BT + (size_t)ncol * C_;
  v8f acc = {};
  FragBF fa, fb;
  for (int kk = 0; kk < 6; ++kk) {
    int ka = kk * 32 + half * 8;    // A lane K base (16-bit 16x32 layout)
    int kb = kk * 32 + half * 8;    // B lane K base (same K order as A)
    const uint4* ap = (const uint4*)(arow + ka);
    const uint4* bp = (const uint4*)(brow + kb);
    fa.u[0] = ap[0]; fa.u[1] = ap[2];   // K {ka..ka+7}, {ka+16..ka+23}
    fb.u[0] = bp[0]; fb.u[1] = bp[2];   // K {kb..kb+7}, {kb+16..kb+23}
    if (kk < 5) {                        // gfx1250 global_prefetch_b8 for next K tile
      __builtin_prefetch(arow + ka + 32, 0, 1);
      __builtin_prefetch(brow + kb + 32, 0, 1);
    }
    acc = __builtin_amdgcn_wmma_f32_16x16x32_bf16(
        /*neg_a=*/false, fa.bf, /*neg_b=*/false, fb.bf,
        /*c_mod=*/(short)0, acc, /*reuse_a=*/false, /*reuse_b=*/false);
  }
  int orow0 = blockIdx.x * 16 + half * 8;
  for (int r = 0; r < 8; ++r) {
    float v = acc[r];
    if (act) v = 1.f / (1.f + __expf(-v));           // fused sigmoid for the r-gate GEMM
    Cout[(size_t)(orow0 + r) * C_ + ncol] = v;
  }
}

// ---------------- kernel 4: ada_peak gates + 16-offset bilinear ring, v += peak ----------------
__global__ void __launch_bounds__(192) k_peak(
    const float* __restrict__ x,
    const float* __restrict__ cvw, const float* __restrict__ cvb,
    const float* __restrict__ bnsv, const float* __restrict__ bnbv,
    const float* __restrict__ chw, const float* __restrict__ chb,
    const float* __restrict__ bnsh, const float* __restrict__ bnbh,
    float* __restrict__ v) {
  __shared__ float red[4][192];
  __shared__ float gate[4];
  int c = threadIdx.x;
  int b = blockIdx.x >> 12;       // / 4096
  int t = blockIdx.x & 4095;
  int h = t >> 6, w = t & 63;
  const float* xb = x + (size_t)b * T_ * C_;
  float s0 = 0, s1 = 0, s2 = 0, s3 = 0;
  for (int j = 0; j < 7; ++j) {
    int hh = h + j - 3;
    if (hh >= 0 && hh < H_) {
      float xv_ = xb[(size_t)(hh * W_ + w) * C_ + c];
      s0 += xv_ * cvw[(0 * C_ + c) * 7 + j];
      s1 += xv_ * cvw[(1 * C_ + c) * 7 + j];
    }
    int ww = w + j - 3;
    if (ww >= 0 && ww < W_) {
      float xh_ = xb[(size_t)(h * W_ + ww) * C_ + c];
      s2 += xh_ * chw[(0 * C_ + c) * 7 + j];
      s3 += xh_ * chw[(1 * C_ + c) * 7 + j];
    }
  }
  red[0][c] = s0; red[1][c] = s1; red[2][c] = s2; red[3][c] = s3;
  __syncthreads();
  if (c < 4) {
    float s = 0.f;
    for (int i = 0; i < C_; ++i) s += red[c][i];
    const float inv = 0.99999500003749968f;          // 1/sqrt(1+1e-5)
    float bias = (c < 2) ? cvb[c]  : chb[c - 2];
    float sc   = (c < 2) ? bnsv[c] : bnsh[c - 2];
    float bb   = (c < 2) ? bnbv[c] : bnbh[c - 2];
    float gz = (s + bias) * (sc * inv) + bb;
    gate[c] = 2.f / (1.f + __expf(-gz));             // sigmoid * (GB_MAX-1)
  }
  __syncthreads();
  float gb0 = gate[0], gb1 = gate[1], gb2 = gate[2], gb3 = gate[3];
  float mxs[4] = { -gb0, 0.f, gb1, 0.f };
  float mys[4] = { 0.f, gb3, 0.f, -gb2 };
  const float PRX[16] = {-2,-2,-2,-2, -2,-1,0,1,  2,2,2,2,  -1,0,1,2};
  const float PRY[16] = {-2,-1,0,1,   2,2,2,2,  -1,0,1,2,  -2,-2,-2,-2};
  float xc = xb[(size_t)t * C_ + c];
  float accp = 0.f;
  for (int n = 0; n < 16; ++n) {
    int blk = n >> 2;
    float px = (float)(h + 2) + PRX[n] + mxs[blk];
    float py = (float)(w + 2) + PRY[n] + mys[blk];
    float flx = floorf(px), fly = floorf(py);
    float qlx = fminf(fmaxf(flx, 0.f), 67.f);
    float qly = fminf(fmaxf(fly, 0.f), 67.f);
    float qrx = fminf(fmaxf(flx + 1.f, 0.f), 67.f);
    float qry = fminf(fmaxf(fly + 1.f, 0.f), 67.f);
    float pxc = fminf(fmaxf(px, 0.f), 67.f);
    float pyc = fminf(fmaxf(py, 0.f), 67.f);
    float glt = (1.f + (qlx - pxc)) * (1.f + (qly - pyc));
    float grb = (1.f - (qrx - pxc)) * (1.f - (qry - pyc));
    int ilx = min(max((int)qlx - 2, 0), H_ - 1);     // edge padding
    int ily = min(max((int)qly - 2, 0), W_ - 1);
    int irx = min(max((int)qrx - 2, 0), H_ - 1);
    int iry = min(max((int)qry - 2, 0), W_ - 1);
    float xlt = xb[(size_t)(ilx * W_ + ily) * C_ + c];
    float xrb = xb[(size_t)(irx * W_ + iry) * C_ + c];
    accp += xc - (glt * xlt + grb * xrb);
  }
  v[((size_t)b * T_ + t) * C_ + c] += accp * (1.f / 16.f);
}

// ---------------- bi_wkv chunked scan ----------------
struct ScanS { float m, a, b; };

__device__ __forceinline__ void stepS(ScanS& s, float w, float kt, float vt) {
  float m2 = fmaxf(s.m - w, kt);
  float e1 = __expf(s.m - w - m2);
  float e2 = __expf(kt - m2);
  s.a = e1 * s.a + e2;
  s.b = e1 * s.b + e2 * vt;
  s.m = m2;
}
__device__ __forceinline__ ScanS applySeg(ScanS in, ScanS seg, float Lw) {
  ScanS o;
  o.m = fmaxf(in.m - Lw, seg.m);
  float e1 = __expf(in.m - Lw - o.m);
  float e2 = __expf(seg.m - o.m);
  o.a = e1 * in.a + e2 * seg.a;
  o.b = e1 * in.b + e2 * seg.b;
  return o;
}

// phase 1: per-chunk summaries (both directions), from empty state
__global__ void __launch_bounds__(192) k_chunksum(
    const float* __restrict__ k, const float* __restrict__ v,
    const float* __restrict__ sdec,
    float* __restrict__ sm, float* __restrict__ sa, float* __restrict__ sb) {
  int c  = threadIdx.x;
  int gz = blockIdx.x;                 // dir*B*NC + b*NC + ch
  int ch  = gz % NC_;
  int b   = (gz / NC_) % B_;
  int dir = gz / (NC_ * B_);
  float w = sdec[c] * (1.f / (float)T_);
  ScanS s = { -1e38f, 0.f, 0.f };
  const float* kb = k + (size_t)b * T_ * C_;
  const float* vb = v + (size_t)b * T_ * C_;
  for (int i = 0; i < LC_; ++i) {
    int t = (dir == 0) ? (ch * LC_ + i) : (ch * LC_ + LC_ - 1 - i);
    size_t o = (size_t)t * C_ + c;
    stepS(s, w, kb[o], vb[o]);
  }
  int idx = ((dir * B_ + b) * C_ + c) * NC_ + ch;
  sm[idx] = s.m; sa[idx] = s.a; sb[idx] = s.b;
}

// phase 2: exclusive prefix over chunk summaries
__global__ void __launch_bounds__(192) k_chunkpre(
    const float* __restrict__ sdec,
    const float* __restrict__ sm, const float* __restrict__ sa, const float* __restrict__ sb,
    float* __restrict__ pm, float* __restrict__ pa, float* __restrict__ pb) {
  int c  = threadIdx.x;
  int gz = blockIdx.x;                 // dir*B + b
  int b = gz % B_;
  int dir = gz / B_;
  float Lw = sdec[c] * (1.f / (float)T_) * (float)LC_;
  ScanS s = { -1e38f, 0.f, 0.f };
  int base = ((dir * B_ + b) * C_ + c) * NC_;
  for (int i = 0; i < NC_; ++i) {
    int ch = (dir == 0) ? i : (NC_ - 1 - i);
    pm[base + ch] = s.m; pa[base + ch] = s.a; pb[base + ch] = s.b;
    ScanS seg = { sm[base + ch], sa[base + ch], sb[base + ch] };
    s = applySeg(s, seg, Lw);
  }
}

// phase 3a: forward per-t states (exclusive) written out, chunk-parallel
__global__ void __launch_bounds__(192) k_fwd(
    const float* __restrict__ k, const float* __restrict__ v,
    const float* __restrict__ sdec,
    const float* __restrict__ pm, const float* __restrict__ pa, const float* __restrict__ pb,
    float* __restrict__ mf, float* __restrict__ af, float* __restrict__ bf) {
  int c  = threadIdx.x;
  int ch = blockIdx.x % NC_;
  int b  = blockIdx.x / NC_;
  float w = sdec[c] * (1.f / (float)T_);
  int base = ((0 * B_ + b) * C_ + c) * NC_;
  ScanS s = { pm[base + ch], pa[base + ch], pb[base + ch] };
  const float* kb = k + (size_t)b * T_ * C_;
  const float* vb = v + (size_t)b * T_ * C_;
  float* mfb = mf + (size_t)b * T_ * C_;
  float* afb = af + (size_t)b * T_ * C_;
  float* bfb = bf + (size_t)b * T_ * C_;
  for (int i = 0; i < LC_; ++i) {
    int t = ch * LC_ + i;
    size_t o = (size_t)t * C_ + c;
    mfb[o] = s.m; afb[o] = s.a; bfb[o] = s.b;
    stepS(s, w, kb[o], vb[o]);
  }
}

// phase 3b: backward states on the fly, combine with stored fwd states -> sr*y (bf16)
__global__ void __launch_bounds__(192) k_bwd(
    const float* __restrict__ k, const float* __restrict__ v, const float* __restrict__ sr,
    const float* __restrict__ sdec, const float* __restrict__ sfir,
    const float* __restrict__ pm, const float* __restrict__ pa, const float* __restrict__ pb,
    const float* __restrict__ mf, const float* __restrict__ af, const float* __restrict__ bf,
    unsigned short* __restrict__ sry) {
  int c  = threadIdx.x;
  int ch = blockIdx.x % NC_;
  int b  = blockIdx.x / NC_;
  float w = sdec[c] * (1.f / (float)T_);
  float u = sfir[c] * (1.f / (float)T_);
  int base = ((1 * B_ + b) * C_ + c) * NC_;
  ScanS s = { pm[base + ch], pa[base + ch], pb[base + ch] };
  const float* kb = k + (size_t)b * T_ * C_;
  const float* vb = v + (size_t)b * T_ * C_;
  const float* srb = sr + (size_t)b * T_ * C_;
  const float* mfb = mf + (size_t)b * T_ * C_;
  const float* afb = af + (size_t)b * T_ * C_;
  const float* bfb = bf + (size_t)b * T_ * C_;
  unsigned short* sryb = sry + (size_t)b * T_ * C_;
  for (int i = LC_ - 1; i >= 0; --i) {
    int t = ch * LC_ + i;
    size_t o = (size_t)t * C_ + c;
    float kt = kb[o], vt = vb[o];
    float ms = u + kt;
    float mft = mfb[o];
    float M = fmaxf(fmaxf(mft, s.m), ms);
    float ef = __expf(mft - M), eb = __expf(s.m - M), es = __expf(ms - M);
    float num = ef * bfb[o] + eb * s.b + es * vt;
    float den = ef * afb[o] + eb * s.a + es;
    float y = num / den;
    sryb[o] = f2bf(srb[o] * y);
    stepS(s, w, kt, vt);
  }
}

// ---------------- host-side launch ----------------
extern "C" void kernel_launch(void* const* d_in, const int* in_sizes, int n_in,
                              void* d_out, int out_size, void* d_ws, size_t ws_size,
                              hipStream_t stream) {
  (void)in_sizes; (void)n_in; (void)out_size; (void)ws_size;
  const float* x    = (const float*)d_in[0];
  const float* mk   = (const float*)d_in[3];
  const float* mv   = (const float*)d_in[4];
  const float* mr   = (const float*)d_in[5];
  const float* Wk   = (const float*)d_in[6];
  const float* Wv   = (const float*)d_in[7];
  const float* Wr   = (const float*)d_in[8];
  const float* Wo   = (const float*)d_in[9];
  const float* sdec = (const float*)d_in[10];
  const float* sfir = (const float*)d_in[11];
  const float* cvw  = (const float*)d_in[12];
  const float* cvb  = (const float*)d_in[13];
  const float* bnsv = (const float*)d_in[14];
  const float* bnbv = (const float*)d_in[15];
  const float* chw  = (const float*)d_in[16];
  const float* chb  = (const float*)d_in[17];
  const float* bnsh = (const float*)d_in[18];
  const float* bnbh = (const float*)d_in[19];
  float* out = (float*)d_out;

  // workspace layout (bytes, 256B-aligned blocks)
  const size_t NE   = (size_t)B_ * T_ * C_;          // 1,572,864
  const size_t BF   = NE * 2;                        // bf16 buffer bytes
  const size_t F32  = NE * 4;                        // f32 buffer bytes
  const size_t WTB  = (size_t)4 * C_ * C_ * 2;       // 294,912
  const size_t SUMN = (size_t)2 * B_ * C_ * NC_;     // 49,152 floats per component
  char* ws = (char*)d_ws;
  unsigned short* xk  = (unsigned short*)(ws);
  unsigned short* xv  = (unsigned short*)(ws + BF);
  unsigned short* xr  = (unsigned short*)(ws + 2 * BF);
  unsigned short* WT  = (unsigned short*)(ws + 3 * BF);
  float* kbuf  = (float*)(ws + 3 * BF + WTB);
  float* vbuf  = (float*)(ws + 3 * BF + WTB + F32);
  float* srbuf = (float*)(ws + 3 * BF + WTB + 2 * F32);
  float* sm    = (float*)(ws + 3 * BF + WTB + 3 * F32);
  float* sa    = sm + SUMN;
  float* sb    = sa + SUMN;
  float* pm    = sb + SUMN;
  float* pa    = pm + SUMN;
  float* pb    = pa + SUMN;
  float* mf    = (float*)((char*)(pb + SUMN));
  float* af    = mf + NE;
  float* bfst  = af + NE;
  unsigned short* sry = (unsigned short*)(bfst + NE);

  // 1) shift + mix -> bf16 activations
  k_prep<<<dim3((unsigned)(NE / 256)), dim3(256), 0, stream>>>(x, mk, mv, mr, xk, xv, xr);
  // 2) weights -> bf16, N-major
  k_wtrans<<<dim3(576), dim3(256), 0, stream>>>(Wk, Wv, Wr, Wo, WT);
  // 3) three WMMA GEMMs: k, v, sigmoid(r)
  dim3 gg((B_ * T_) / 16, C_ / 16);                  // 512 x 12 tiles, one wave each
  k_gemm<<<gg, dim3(32), 0, stream>>>(xk, WT + 0 * C_ * C_, kbuf, 0);
  k_gemm<<<gg, dim3(32), 0, stream>>>(xv, WT + 1 * C_ * C_, vbuf, 0);
  k_gemm<<<gg, dim3(32), 0, stream>>>(xr, WT + 2 * C_ * C_, srbuf, 1);
  // 4) ada_peak: gates + ring gather, v += peak
  k_peak<<<dim3(B_ * T_), dim3(C_), 0, stream>>>(x, cvw, cvb, bnsv, bnbv, chw, chb, bnsh, bnbh, vbuf);
  // 5) chunked bidirectional wkv scan
  k_chunksum<<<dim3(2 * B_ * NC_), dim3(C_), 0, stream>>>(kbuf, vbuf, sdec, sm, sa, sb);
  k_chunkpre<<<dim3(2 * B_), dim3(C_), 0, stream>>>(sdec, sm, sa, sb, pm, pa, pb);
  k_fwd<<<dim3(B_ * NC_), dim3(C_), 0, stream>>>(kbuf, vbuf, sdec, pm, pa, pb, mf, af, bfst);
  k_bwd<<<dim3(B_ * NC_), dim3(C_), 0, stream>>>(kbuf, vbuf, srbuf, sdec, sfir,
                                                 pm, pa, pb, mf, af, bfst, sry);
  // 6) final WMMA GEMM: (sr*y) @ Wo -> d_out (f32)
  k_gemm<<<gg, dim3(32), 0, stream>>>(sry, WT + 3 * C_ * C_, out, 0);
}
